// HeteroToolGNN_55465207661146
// MI455X (gfx1250) — compile-verified, matrix operand order checked by reading, not run on hardware
//
#include <hip/hip_runtime.h>

typedef __attribute__((ext_vector_type(16))) __bf16 v16bf;
typedef __attribute__((ext_vector_type(8)))  __bf16 v8bf;
typedef __attribute__((ext_vector_type(8)))  float  v8f;
typedef int v4i_gcc __attribute__((vector_size(4 * sizeof(int))));

constexpr int kNT    = 4;
constexpr int kNPT   = 1024;
constexpr int kIN    = 21;
constexpr int kH     = 256;
constexpr int kL     = 3;
constexpr int kKE    = 48;     // edge types
constexpr int kE     = 8192;   // edges per type
constexpr int kTok   = kNT * kNPT;  // 4096
constexpr int kHeads = 8;
constexpr int kDh    = 32;
constexpr int kET    = 3;

constexpr int BM = 128, BN = 128, BK = 32;
constexpr int LSTR = 40;   // padded LDS row stride (bf16 elems); chunks at +0/+8/+16/+24 stay 16B aligned

// ---------------------------------------------------------------- async global->LDS (CDNA5), guarded
#if defined(__has_builtin)
#if __has_builtin(__builtin_amdgcn_global_load_async_to_lds_b128) && \
    __has_builtin(__builtin_amdgcn_s_wait_asynccnt)
#define USE_ASYNC_LDS 1
#endif
#endif
#ifndef USE_ASYNC_LDS
#define USE_ASYNC_LDS 0
#endif

__device__ inline void cp16(const __bf16* g, __bf16* l) {
#if USE_ASYNC_LDS
  // signature (from hipcc diagnostic): (v4i __device__*, v4i __shared__*, imm offset, imm cpol)
  // generic -> AS-qualified pointer conversions are implicit in HIP mode.
  __builtin_amdgcn_global_load_async_to_lds_b128((v4i_gcc*)g, (v4i_gcc*)l, 0, 0);
#else
  *(uint4*)l = *(const uint4*)g;
#endif
}

template <int N>
__device__ inline void wait_cp() {
#if USE_ASYNC_LDS
  __builtin_amdgcn_s_wait_asynccnt(N);
#endif
}

// ---------------------------------------------------------------- WMMA helpers
__device__ inline v8f wmma_bf16(v16bf a, v16bf b, v8f c) {
  return __builtin_amdgcn_wmma_f32_16x16x32_bf16(false, a, false, b, (short)0, c, false, false);
}

__device__ inline v16bf make_frag(const __bf16* p0, const __bf16* p1) {
  v8bf lo = *(const v8bf*)p0;
  v8bf hi = *(const v8bf*)p1;
  v16bf r;
#pragma unroll
  for (int j = 0; j < 8; ++j) { r[j] = lo[j]; r[8 + j] = hi[j]; }
  return r;
}

__device__ inline float block_sum(float v, float* red) {
  const int tid = threadIdx.x;
  __syncthreads();
  red[tid] = v; __syncthreads();
  for (int s = 128; s > 0; s >>= 1) { if (tid < s) red[tid] += red[tid + s]; __syncthreads(); }
  return red[0];
}

// ---------------------------------------------------------------- small utils
__global__ void zero_f32_kernel(float* p, long n) {
  long i = (long)blockIdx.x * blockDim.x + threadIdx.x;
  if (i < n) p[i] = 0.0f;
}

// fp32 [mat][K][N] -> bf16 [mat][N][K]  (weights: one-time convert + transpose)
__global__ void convert_transpose_kernel(const float* __restrict__ src, __bf16* __restrict__ dst,
                                         int Kd, int Nd, long total) {
  long i = (long)blockIdx.x * blockDim.x + threadIdx.x;
  if (i >= total) return;
  const long kn = (long)Kd * Nd;
  const long mat = i / kn, rem = i - mat * kn;
  const int n = (int)(rem / Kd), k = (int)(rem % Kd);
  dst[i] = (__bf16)src[mat * kn + (long)k * Nd + n];
}

__global__ void edge_count_kernel(const int* __restrict__ ei, float* __restrict__ cnt) {
  const int k = blockIdx.x;
  const int* dst = ei + ((size_t)k * 2 + 1) * kE;
  for (int e = threadIdx.x; e < kE; e += blockDim.x)
    atomicAdd(&cnt[k * kNPT + dst[e]], 1.0f);
}

__global__ void invcnt_kernel(float* c, long n) {
  long i = (long)blockIdx.x * blockDim.x + threadIdx.x;
  if (i < n) c[i] = 1.0f / fmaxf(c[i], 1.0f);
}

// x[t,n,h] = sum_i x_nodes[t,n,i]*proj_W[t,i,h] + proj_b[t,h]   (fp32 + bf16 mirror)
__global__ __launch_bounds__(256) void proj_kernel(const float* __restrict__ xn,
                                                   const float* __restrict__ W,
                                                   const float* __restrict__ b,
                                                   float* __restrict__ x,
                                                   __bf16* __restrict__ xb) {
  __shared__ float xin[kIN];
  const int row = blockIdx.x, tid = threadIdx.x;
  const int t = row / kNPT;
  if (tid < kIN) xin[tid] = xn[(size_t)row * kIN + tid];
  __syncthreads();
  float s = b[t * kH + tid];
  for (int i = 0; i < kIN; ++i) s += xin[i] * W[((size_t)t * kIN + i) * kH + tid];
  x[(size_t)row * kH + tid] = s;
  xb[(size_t)row * kH + tid] = (__bf16)s;
}

// x = relu(LN(acc + x; g[type], b[type]))  (fp32 + bf16 mirror)
__global__ __launch_bounds__(256) void combine_ln_relu_kernel(const float* __restrict__ acc,
                                                              float* __restrict__ x,
                                                              __bf16* __restrict__ xb,
                                                              const float* __restrict__ g,
                                                              const float* __restrict__ b) {
  __shared__ float red[256];
  const int row = blockIdx.x, tid = threadIdx.x;
  const int t = row / kNPT;
  float v = acc[(size_t)row * kH + tid] + x[(size_t)row * kH + tid];
  float mean = block_sum(v, red) * (1.0f / kH);
  float d = v - mean;
  float var = block_sum(d * d, red) * (1.0f / kH);
  float y = fmaxf(d * rsqrtf(var + 1e-5f) * g[t * kH + tid] + b[t * kH + tid], 0.0f);
  x[(size_t)row * kH + tid] = y;
  xb[(size_t)row * kH + tid] = (__bf16)y;
}

// ---------------------------------------------------------------- conv GEMM (WMMA, async double-buffered staging,
//                                                                  fused gather + scatter-mean epilogue)
__global__ __launch_bounds__(256) void conv_gemm_kernel(const __bf16* __restrict__ xb,
                                                        const __bf16* __restrict__ WTl,  // [48][N=256][K=256]
                                                        const float* __restrict__ bl,    // [48][256]
                                                        const int* __restrict__ ei,
                                                        const float* __restrict__ invcnt,
                                                        float* __restrict__ out) {
  __shared__ __align__(16) __bf16 As[2][BM][LSTR];
  __shared__ __align__(16) __bf16 Bs[2][BN][LSTR];
  __shared__ int   sidx[BM];
  __shared__ int   sdst[BM];
  __shared__ float sscale[BM];

  const int k  = blockIdx.y;
  const int st = k / (kNT * kET);
  const int dt = (k / kET) & (kNT - 1);
  const int* srcp = ei + ((size_t)k * 2 + 0) * kE;
  const int* dstp = ei + ((size_t)k * 2 + 1) * kE;
  const __bf16* BT = WTl + (size_t)k * kH * kH;  // [n][kk]
  const int m0 = blockIdx.x * BM;
  const int n0 = blockIdx.z * BN;

  const int tid = threadIdx.x, lane = tid & 31, wave = tid >> 5;
  const int wm = wave & 3, wn = wave >> 2;
  const int half = lane >> 4, l16 = lane & 15, kb8 = half * 8;

  if (tid < BM) {
    const int e = m0 + tid;
    const int s = srcp[e], d = dstp[e];
    sidx[tid] = s; sdst[tid] = d;
    sscale[tid] = invcnt[k * kNPT + d];
  }
  __syncthreads();

  // 1024 16B-chunks per tile (512 A + 512 B) -> 4 async copies per thread
  auto stage = [&](int buf, int k0) {
#pragma unroll
    for (int c = tid; c < 1024; c += 256) {
      if (c < 512) {
        const int r = c >> 2, ch = c & 3;
        cp16(xb + ((size_t)(st * kNPT) + sidx[r]) * kH + k0 + ch * 8, &As[buf][r][ch * 8]);
      } else {
        const int n = (c - 512) >> 2, ch = c & 3;
        cp16(BT + (size_t)(n0 + n) * kH + k0 + ch * 8, &Bs[buf][n][ch * 8]);
      }
    }
  };

  v8f acc[2][4];
#pragma unroll
  for (int mi = 0; mi < 2; ++mi)
#pragma unroll
    for (int ni = 0; ni < 4; ++ni)
#pragma unroll
      for (int r = 0; r < 8; ++r) acc[mi][ni][r] = 0.0f;

  constexpr int TKT = kH / BK;  // 8
  stage(0, 0);
  for (int t = 0; t < TKT; ++t) {
    const int buf = t & 1;
    if (t + 1 < TKT) { stage(buf ^ 1, (t + 1) * BK); wait_cp<4>(); }
    else             { wait_cp<0>(); }
    __syncthreads();

    v16bf afr[2], bfr[4];
#pragma unroll
    for (int mi = 0; mi < 2; ++mi) {
      const int m = wm * 32 + mi * 16 + l16;
      afr[mi] = make_frag(&As[buf][m][kb8], &As[buf][m][16 + kb8]);
    }
#pragma unroll
    for (int ni = 0; ni < 4; ++ni) {
      const int n = wn * 64 + ni * 16 + l16;
      bfr[ni] = make_frag(&Bs[buf][n][kb8], &Bs[buf][n][16 + kb8]);
    }
#pragma unroll
    for (int mi = 0; mi < 2; ++mi)
#pragma unroll
      for (int ni = 0; ni < 4; ++ni)
        acc[mi][ni] = wmma_bf16(afr[mi], bfr[ni], acc[mi][ni]);
    __syncthreads();
  }

  // fused segment-mean + cross-edge-type sum: atomicAdd((msg+bias)*inv_cnt)
#pragma unroll
  for (int mi = 0; mi < 2; ++mi)
#pragma unroll
    for (int ni = 0; ni < 4; ++ni) {
      const int col = n0 + wn * 64 + ni * 16 + l16;
      const float bv = bl[k * kH + col];
#pragma unroll
      for (int r = 0; r < 8; ++r) {
        const int rl = wm * 32 + mi * 16 + half * 8 + r;
        const int dn = sdst[rl];
        const float sc = sscale[rl];
        atomicAdd(&out[((size_t)dt * kNPT + dn) * kH + col], (acc[mi][ni][r] + bv) * sc);
      }
    }
}

// ---------------------------------------------------------------- generic GEMM (WMMA): C = A(bf16) @ B + bias
// B supplied pre-transposed bf16 [mat][Ntot][Kdim]
__global__ __launch_bounds__(256) void gemm_bias_kernel(const __bf16* __restrict__ A, int Kdim, int Ntot,
                                                        const __bf16* __restrict__ BTall, long strideB,
                                                        const float* __restrict__ biasAll, long strideBias,
                                                        void* __restrict__ Call, long strideC, int out_bf16) {
  __shared__ __align__(16) __bf16 As[2][BM][LSTR];
  __shared__ __align__(16) __bf16 Bs[2][BN][LSTR];
  const int mat = blockIdx.y;
  const __bf16* BT = BTall + (size_t)mat * strideB;
  const float* bias = biasAll + (size_t)mat * strideBias;
  const int m0 = blockIdx.x * BM;
  const int n0 = blockIdx.z * BN;

  const int tid = threadIdx.x, lane = tid & 31, wave = tid >> 5;
  const int wm = wave & 3, wn = wave >> 2;
  const int half = lane >> 4, l16 = lane & 15, kb8 = half * 8;

  auto stage = [&](int buf, int k0) {
#pragma unroll
    for (int c = tid; c < 1024; c += 256) {
      if (c < 512) {
        const int r = c >> 2, ch = c & 3;
        cp16(A + (size_t)(m0 + r) * Kdim + k0 + ch * 8, &As[buf][r][ch * 8]);
      } else {
        const int n = (c - 512) >> 2, ch = c & 3;
        cp16(BT + (size_t)(n0 + n) * Kdim + k0 + ch * 8, &Bs[buf][n][ch * 8]);
      }
    }
  };

  v8f acc[2][4];
#pragma unroll
  for (int mi = 0; mi < 2; ++mi)
#pragma unroll
    for (int ni = 0; ni < 4; ++ni)
#pragma unroll
      for (int r = 0; r < 8; ++r) acc[mi][ni][r] = 0.0f;

  const int TKT = Kdim / BK;
  stage(0, 0);
  for (int t = 0; t < TKT; ++t) {
    const int buf = t & 1;
    if (t + 1 < TKT) { stage(buf ^ 1, (t + 1) * BK); wait_cp<4>(); }
    else             { wait_cp<0>(); }
    __syncthreads();

    v16bf afr[2], bfr[4];
#pragma unroll
    for (int mi = 0; mi < 2; ++mi) {
      const int m = wm * 32 + mi * 16 + l16;
      afr[mi] = make_frag(&As[buf][m][kb8], &As[buf][m][16 + kb8]);
    }
#pragma unroll
    for (int ni = 0; ni < 4; ++ni) {
      const int n = wn * 64 + ni * 16 + l16;
      bfr[ni] = make_frag(&Bs[buf][n][kb8], &Bs[buf][n][16 + kb8]);
    }
#pragma unroll
    for (int mi = 0; mi < 2; ++mi)
#pragma unroll
      for (int ni = 0; ni < 4; ++ni)
        acc[mi][ni] = wmma_bf16(afr[mi], bfr[ni], acc[mi][ni]);
    __syncthreads();
  }

#pragma unroll
  for (int mi = 0; mi < 2; ++mi)
#pragma unroll
    for (int ni = 0; ni < 4; ++ni) {
      const int col = n0 + wn * 64 + ni * 16 + l16;
      const float bv = bias[col];
#pragma unroll
      for (int r = 0; r < 8; ++r) {
        const int row = m0 + wm * 32 + mi * 16 + half * 8 + r;
        const float v = acc[mi][ni][r] + bv;
        if (out_bf16)
          ((__bf16*)Call)[(size_t)mat * strideC + (size_t)row * Ntot + col] = (__bf16)v;
        else
          ((float*)Call)[(size_t)mat * strideC + (size_t)row * Ntot + col] = v;
      }
    }
}

// ---------------------------------------------------------------- flash attention (WMMA, streaming softmax)
__global__ __launch_bounds__(256) void flash_attn_kernel(const __bf16* __restrict__ qb,
                                                         const __bf16* __restrict__ kbuf,
                                                         const __bf16* __restrict__ vbuf,
                                                         __bf16* __restrict__ out) {
  __shared__ __align__(16) __bf16 Qs[128][LSTR];
  __shared__ __align__(16) __bf16 Kt[32][LSTR];
  __shared__ __align__(16) __bf16 Vt[32][LSTR];
  __shared__ __align__(16) __bf16 Ps[8][16][32];

  const int h = blockIdx.y;
  const int q0 = blockIdx.x * 128;
  const int tid = threadIdx.x, lane = tid & 31, wave = tid >> 5;
  const int half = lane >> 4, l16 = lane & 15, kb8 = half * 8;

#pragma unroll
  for (int c = tid; c < 512; c += 256) {  // Q tile: 128 rows x 64B
    const int r = c >> 2, ch = c & 3;
    cp16(qb + (size_t)(q0 + r) * kH + h * kDh + ch * 8, &Qs[r][ch * 8]);
  }
  wait_cp<0>();
  __syncthreads();
  v16bf qa;
  {
    const int m = wave * 16 + l16;
    qa = make_frag(&Qs[m][kb8], &Qs[m][16 + kb8]);
  }

  v8f o0, o1;
  float mrow[8], lrow[8];
#pragma unroll
  for (int r = 0; r < 8; ++r) { o0[r] = 0.0f; o1[r] = 0.0f; mrow[r] = -1e30f; lrow[r] = 0.0f; }
  const float scale = 0.17677669529663687f;  // 1/sqrt(32)

  for (int kt = 0; kt < kTok; kt += 32) {
    __syncthreads();
    {  // K/V tiles: 2 * 32 rows x 64B = 256 chunks, one per thread
      if (tid < 128) {
        const int r = tid >> 2, ch = tid & 3;
        cp16(kbuf + (size_t)(kt + r) * kH + h * kDh + ch * 8, &Kt[r][ch * 8]);
      } else {
        const int c2 = tid - 128, r = c2 >> 2, ch = c2 & 3;
        cp16(vbuf + (size_t)(kt + r) * kH + h * kDh + ch * 8, &Vt[r][ch * 8]);
      }
    }
    wait_cp<0>();
    __syncthreads();

    // scores (K = Dh = 32 -> one WMMA per 16-key tile)
    v16bf bk0 = make_frag(&Kt[l16][kb8], &Kt[l16][16 + kb8]);
    v16bf bk1 = make_frag(&Kt[16 + l16][kb8], &Kt[16 + l16][16 + kb8]);
    v8f z;
#pragma unroll
    for (int r = 0; r < 8; ++r) z[r] = 0.0f;
    v8f s0 = wmma_bf16(qa, bk0, z);
    v8f s1 = wmma_bf16(qa, bk1, z);
#pragma unroll
    for (int r = 0; r < 8; ++r) { s0[r] *= scale; s1[r] *= scale; }

    v8f mx;
#pragma unroll
    for (int r = 0; r < 8; ++r) mx[r] = fmaxf(s0[r], s1[r]);
    for (int d = 1; d < 16; d <<= 1)
#pragma unroll
      for (int r = 0; r < 8; ++r) mx[r] = fmaxf(mx[r], __shfl_xor(mx[r], d, 32));

    float alpha[8];
#pragma unroll
    for (int r = 0; r < 8; ++r) {
      const float mn = fmaxf(mrow[r], mx[r]);
      alpha[r] = __expf(mrow[r] - mn);
      mrow[r] = mn;
      s0[r] = __expf(s0[r] - mn);
      s1[r] = __expf(s1[r] - mn);
    }
    v8f rs;
#pragma unroll
    for (int r = 0; r < 8; ++r) rs[r] = s0[r] + s1[r];
    for (int d = 1; d < 16; d <<= 1)
#pragma unroll
      for (int r = 0; r < 8; ++r) rs[r] += __shfl_xor(rs[r], d, 32);
#pragma unroll
    for (int r = 0; r < 8; ++r) {
      lrow[r] = lrow[r] * alpha[r] + rs[r];
      o0[r] *= alpha[r];
      o1[r] *= alpha[r];
    }

    // restage P (C-layout -> A-fragment layout) through per-wave LDS patch
#pragma unroll
    for (int r = 0; r < 8; ++r) {
      Ps[wave][half * 8 + r][l16] = (__bf16)s0[r];
      Ps[wave][half * 8 + r][16 + l16] = (__bf16)s1[r];
    }
    __syncthreads();
    v16bf pa = make_frag(&Ps[wave][l16][kb8], &Ps[wave][l16][16 + kb8]);

    v16bf bv0, bv1;
#pragma unroll
    for (int j = 0; j < 8; ++j) {
      bv0[j]     = Vt[kb8 + j][l16];
      bv0[8 + j] = Vt[16 + kb8 + j][l16];
      bv1[j]     = Vt[kb8 + j][16 + l16];
      bv1[8 + j] = Vt[16 + kb8 + j][16 + l16];
    }
    o0 = wmma_bf16(pa, bv0, o0);
    o1 = wmma_bf16(pa, bv1, o1);
  }

#pragma unroll
  for (int r = 0; r < 8; ++r) {
    const float inv = 1.0f / lrow[r];
    const int qrow = q0 + wave * 16 + half * 8 + r;
    out[(size_t)qrow * kH + h * kDh + l16]      = (__bf16)(o0[r] * inv);
    out[(size_t)qrow * kH + h * kDh + 16 + l16] = (__bf16)(o1[r] * inv);
  }
}

// ---------------------------------------------------------------- pooling + tiny heads
__global__ __launch_bounds__(256) void pool_logits_kernel(const float* __restrict__ z,
                                                          const float* __restrict__ pw,
                                                          const float* __restrict__ pb,
                                                          float* __restrict__ logits) {
  __shared__ float red[256];
  const int row = blockIdx.x, tid = threadIdx.x;
  const float s = block_sum(z[(size_t)row * kH + tid] * pw[tid], red);
  if (tid == 0) logits[row] = s + pb[0];
}

__global__ __launch_bounds__(256) void softmax_w_kernel(const float* __restrict__ logits,
                                                        float* __restrict__ w) {
  __shared__ float red[256];
  const int tid = threadIdx.x;
  float m = -1e30f;
  for (int i = tid; i < kTok; i += 256) m = fmaxf(m, logits[i]);
  red[tid] = m; __syncthreads();
  for (int s = 128; s > 0; s >>= 1) { if (tid < s) red[tid] = fmaxf(red[tid], red[tid + s]); __syncthreads(); }
  m = red[0]; __syncthreads();
  float s = 0.0f;
  for (int i = tid; i < kTok; i += 256) s += __expf(logits[i] - m);
  s = block_sum(s, red);
  const float inv = 1.0f / s;
  for (int i = tid; i < kTok; i += 256) w[i] = __expf(logits[i] - m) * inv;
}

__global__ __launch_bounds__(256) void emb_kernel(const float* __restrict__ z,
                                                  const float* __restrict__ w,
                                                  const float* __restrict__ cx,
                                                  const float* __restrict__ cW,
                                                  const float* __restrict__ cb,
                                                  float* __restrict__ emb) {
  const int h = threadIdx.x;
  float s = 0.0f;
  for (int n = 0; n < kTok; ++n) s += w[n] * z[(size_t)n * kH + h];
  for (int c = 0; c < 32; ++c) s += cx[c] * cW[c * kH + h];
  emb[h] = s + cb[h];
}

__global__ __launch_bounds__(256) void heads_kernel(
    const float* __restrict__ emb,
    const float* __restrict__ ta_W1, const float* __restrict__ ta_b1,
    const float* __restrict__ ta_g,  const float* __restrict__ ta_beta,
    const float* __restrict__ ta_W2, const float* __restrict__ ta_b2,
    const float* __restrict__ sh_W1, const float* __restrict__ sh_b1,
    const float* __restrict__ sh_W2, const float* __restrict__ sh_b2,
    const float* __restrict__ fpr_Wa, const float* __restrict__ fpr_ba,
    const float* __restrict__ fpr_g,  const float* __restrict__ fpr_beta,
    const float* __restrict__ fpr_W1, const float* __restrict__ fpr_b1,
    const float* __restrict__ fpr_W2, const float* __restrict__ fpr_b2,
    const float* __restrict__ sec_W1, const float* __restrict__ sec_b1,
    const float* __restrict__ sec_W2, const float* __restrict__ sec_b2,
    float* __restrict__ outp) {
  __shared__ float e[kH], t1[kH], t2[kH], secE[kH], fprE[kH], hh[128], red[256], res[25];
  const int tid = threadIdx.x;
  e[tid] = emb[tid];
  __syncthreads();

  // tool attention: i=0 -> sec_emb, i=1 -> fpr_emb
  for (int i = 0; i < 2; ++i) {
    float a = ta_b1[i * kH + tid];
    for (int j = 0; j < kH; ++j) a += e[j] * ta_W1[((size_t)i * kH + j) * kH + tid];
    t1[tid] = a; __syncthreads();
    const float mean = block_sum(t1[tid], red) * (1.0f / kH);
    const float d = t1[tid] - mean;
    const float var = block_sum(d * d, red) * (1.0f / kH);
    t2[tid] = fmaxf(d * rsqrtf(var + 1e-5f) * ta_g[i * kH + tid] + ta_beta[i * kH + tid], 0.0f);
    __syncthreads();
    float a2 = ta_b2[i * kH + tid];
    for (int j = 0; j < kH; ++j) a2 += t2[j] * ta_W2[((size_t)i * kH + j) * kH + tid];
    const float v = e[tid] * (1.0f / (1.0f + __expf(-a2)));
    if (i == 0) secE[tid] = v; else fprE[tid] = v;
    __syncthreads();
  }

  // simple heads on emb: tpr(0..4), acc(10..14), prec(15..19), rec(20..24)
  const int shBase[4] = {0, 10, 15, 20};
  for (int i = 0; i < 4; ++i) {
    if (tid < 128) {
      float a = sh_b1[i * 128 + tid];
      for (int j = 0; j < kH; ++j) a += e[j] * sh_W1[((size_t)i * kH + j) * 128 + tid];
      hh[tid] = fmaxf(a, 0.0f);
    }
    __syncthreads();
    if (tid < 5) {
      float a = sh_b2[i * 5 + tid];
      for (int j = 0; j < 128; ++j) a += hh[j] * sh_W2[((size_t)i * 128 + j) * 5 + tid];
      res[shBase[i] + tid] = 1.0f / (1.0f + __expf(-a));
    }
    __syncthreads();
  }

  // fpr chain on fpr_emb -> res[5..9]
  {
    float a = fpr_ba[tid];
    for (int j = 0; j < kH; ++j) a += fprE[j] * fpr_Wa[(size_t)j * kH + tid];
    t1[tid] = a; __syncthreads();
    const float mean = block_sum(t1[tid], red) * (1.0f / kH);
    const float d = t1[tid] - mean;
    const float var = block_sum(d * d, red) * (1.0f / kH);
    t2[tid] = fmaxf(d * rsqrtf(var + 1e-5f) * fpr_g[tid] + fpr_beta[tid], 0.0f);
    __syncthreads();
    if (tid < 128) {
      float a1 = fpr_b1[tid];
      for (int j = 0; j < kH; ++j) a1 += t2[j] * fpr_W1[(size_t)j * 128 + tid];
      hh[tid] = fmaxf(a1, 0.0f);
    }
    __syncthreads();
    if (tid < 5) {
      float a2 = fpr_b2[tid];
      for (int j = 0; j < 128; ++j) a2 += hh[j] * fpr_W2[(size_t)j * 5 + tid];
      res[5 + tid] = 1.0f / (1.0f + __expf(-a2));
    }
    __syncthreads();
  }

  // sec heads on sec_emb overwrite index SEC_IDX=1 of each output
  const int secSlot[5] = {1, 6, 11, 16, 21};
  for (int i = 0; i < 5; ++i) {
    if (tid < 128) {
      float a = sec_b1[i * 128 + tid];
      for (int j = 0; j < kH; ++j) a += secE[j] * sec_W1[((size_t)i * kH + j) * 128 + tid];
      hh[tid] = fmaxf(a, 0.0f);
    }
    __syncthreads();
    if (tid == 0) {
      float a = sec_b2[i];
      for (int j = 0; j < 128; ++j) a += hh[j] * sec_W2[(size_t)i * 128 + j];
      res[secSlot[i]] = 1.0f / (1.0f + __expf(-a));
    }
    __syncthreads();
  }
  if (tid < 25) outp[tid] = res[tid];
}

// ---------------------------------------------------------------- launch
extern "C" void kernel_launch(void* const* d_in, const int* in_sizes, int n_in,
                              void* d_out, int out_size, void* d_ws, size_t ws_size,
                              hipStream_t stream) {
  (void)in_sizes; (void)n_in; (void)out_size; (void)ws_size;
  const float* x_nodes    = (const float*)d_in[0];
  const float* contract_x = (const float*)d_in[1];
  const float* proj_W     = (const float*)d_in[2];
  const float* proj_b     = (const float*)d_in[3];
  const float* conv_W     = (const float*)d_in[4];
  const float* conv_b     = (const float*)d_in[5];
  const float* norm_g     = (const float*)d_in[6];
  const float* norm_b     = (const float*)d_in[7];
  const float* attn_W     = (const float*)d_in[8];
  const float* attn_b     = (const float*)d_in[9];
  const float* pool_W     = (const float*)d_in[10];
  const float* pool_b     = (const float*)d_in[11];
  const float* contract_W = (const float*)d_in[12];
  const float* contract_b = (const float*)d_in[13];
  const float* ta_W1      = (const float*)d_in[14];
  const float* ta_b1      = (const float*)d_in[15];
  const float* ta_g       = (const float*)d_in[16];
  const float* ta_beta    = (const float*)d_in[17];
  const float* ta_W2      = (const float*)d_in[18];
  const float* ta_b2      = (const float*)d_in[19];
  const float* sh_W1      = (const float*)d_in[20];
  const float* sh_b1      = (const float*)d_in[21];
  const float* sh_W2      = (const float*)d_in[22];
  const float* sh_b2      = (const float*)d_in[23];
  const float* fpr_Wa     = (const float*)d_in[24];
  const float* fpr_ba     = (const float*)d_in[25];
  const float* fpr_g      = (const float*)d_in[26];
  const float* fpr_beta   = (const float*)d_in[27];
  const float* fpr_W1     = (const float*)d_in[28];
  const float* fpr_b1     = (const float*)d_in[29];
  const float* fpr_W2     = (const float*)d_in[30];
  const float* fpr_b2     = (const float*)d_in[31];
  const float* sec_W1     = (const float*)d_in[32];
  const float* sec_b1     = (const float*)d_in[33];
  const float* sec_W2     = (const float*)d_in[34];
  const float* sec_b2     = (const float*)d_in[35];
  const int*   edge_index = (const int*)d_in[36];

  char* ws = (char*)d_ws;
  size_t off = 0;
  auto take = [&](size_t bytes) -> char* {
    char* p = ws + off;
    off = (off + bytes + 255) & ~(size_t)255;
    return p;
  };
  const size_t HH = (size_t)kH * kH;
  float*  xbuf    = (float*)take((size_t)kTok * kH * 4);
  __bf16* xb      = (__bf16*)take((size_t)kTok * kH * 2);
  float*  xacc    = (float*)take((size_t)kTok * kH * 4);
  __bf16* convWbT = (__bf16*)take((size_t)kL * kKE * HH * 2);  // [l][k][n][kk]
  __bf16* attnWbT = (__bf16*)take((size_t)4 * HH * 2);         // [mat][n][kk]
  float*  invcnt  = (float*)take((size_t)kKE * kNPT * 4);
  __bf16* qkvb    = (__bf16*)take((size_t)3 * kTok * kH * 2);
  __bf16* attnout = (__bf16*)take((size_t)kTok * kH * 2);
  float*  z2      = (float*)take((size_t)kTok * kH * 4);
  float*  logits  = (float*)take((size_t)kTok * 4);
  float*  wvec    = (float*)take((size_t)kTok * 4);
  float*  embv    = (float*)take((size_t)kH * 4);

  // one-time prep: weight convert+transpose, edge counts
  {
    long n = (long)kL * kKE * HH;
    convert_transpose_kernel<<<(n + 255) / 256, 256, 0, stream>>>(conv_W, convWbT, kH, kH, n);
  }
  {
    long n = 4L * HH;
    convert_transpose_kernel<<<(n + 255) / 256, 256, 0, stream>>>(attn_W, attnWbT, kH, kH, n);
  }
  {
    long n = (long)kKE * kNPT;
    zero_f32_kernel<<<(n + 255) / 256, 256, 0, stream>>>(invcnt, n);
    edge_count_kernel<<<kKE, 256, 0, stream>>>(edge_index, invcnt);
    invcnt_kernel<<<(n + 255) / 256, 256, 0, stream>>>(invcnt, n);
  }

  proj_kernel<<<kTok, 256, 0, stream>>>(x_nodes, proj_W, proj_b, xbuf, xb);

  for (int l = 0; l < kL; ++l) {
    long n = (long)kTok * kH;
    zero_f32_kernel<<<(n + 255) / 256, 256, 0, stream>>>(xacc, n);
    dim3 g(kE / BM, kKE, kH / BN);
    conv_gemm_kernel<<<g, 256, 0, stream>>>(xb, convWbT + (size_t)l * kKE * HH,
                                            conv_b + (size_t)l * kKE * kH,
                                            edge_index, invcnt, xacc);
    combine_ln_relu_kernel<<<kTok, 256, 0, stream>>>(xacc, xbuf, xb,
                                                     norm_g + (size_t)l * kNT * kH,
                                                     norm_b + (size_t)l * kNT * kH);
  }

  {  // q, k, v batched (bf16 outputs)
    dim3 g(kTok / BM, 3, kH / BN);
    gemm_bias_kernel<<<g, 256, 0, stream>>>(xb, kH, kH, attnWbT, (long)HH, attn_b, kH,
                                            qkvb, (long)kTok * kH, 1);
  }
  {
    dim3 g(kTok / 128, kHeads, 1);
    flash_attn_kernel<<<g, 256, 0, stream>>>(qkvb, qkvb + (size_t)kTok * kH,
                                             qkvb + (size_t)2 * kTok * kH, attnout);
  }
  {  // output projection (bf16 A)
    dim3 g(kTok / BM, 1, kH / BN);
    gemm_bias_kernel<<<g, 256, 0, stream>>>(attnout, kH, kH, attnWbT + 3 * HH, 0,
                                            attn_b + 3 * kH, 0, z2, 0, 0);
  }

  pool_logits_kernel<<<kTok, 256, 0, stream>>>(z2, pool_W, pool_b, logits);
  softmax_w_kernel<<<1, 256, 0, stream>>>(logits, wvec);
  emb_kernel<<<1, 256, 0, stream>>>(z2, wvec, contract_x, contract_W, contract_b, embv);
  heads_kernel<<<1, 256, 0, stream>>>(embv, ta_W1, ta_b1, ta_g, ta_beta, ta_W2, ta_b2,
                                      sh_W1, sh_b1, sh_W2, sh_b2,
                                      fpr_Wa, fpr_ba, fpr_g, fpr_beta,
                                      fpr_W1, fpr_b1, fpr_W2, fpr_b2,
                                      sec_W1, sec_b1, sec_W2, sec_b2,
                                      (float*)d_out);
}
